// Stage1_63299228008584
// MI455X (gfx1250) — matrix operand order repaired and checked by reading, compiled, not run on hardware
//
#include <hip/hip_runtime.h>
#include <hip/hip_bf16.h>
#include <stdint.h>

// ---------------------------------------------------------------------------
// MI455X (gfx1250) implementation.
// Conv stack = ~120 GFLOP, memory traffic only ~200MB -> compute bound:
// use v_wmma_f32_16x16x32_bf16 (bf16 in, f32 accumulate) for layers 2-4
// via implicit GEMM over NHWC bf16 activations. Layer 1 (Cin=3) is direct
// VALU conv fused with its 2x2 maxpool. Anchor/IoU targets are tiny VALU.
// All conv dims are compile-time (powers of two) so index math is shifts.
// Wave tile = 16M x 64N: one A fragment feeds 4 WMMAs (A-load amortized).
// ---------------------------------------------------------------------------

typedef __attribute__((ext_vector_type(16))) __bf16 v16bf;
typedef __attribute__((ext_vector_type(8)))  float  v8f;

struct alignas(16) U128 { unsigned int x, y, z, w; };
union Frag16 { v16bf v; U128 q[2]; };

__device__ __forceinline__ unsigned short f2bf(float f) {
    unsigned int u = __float_as_uint(f);
    u = u + 0x7FFFu + ((u >> 16) & 1u);          // round-to-nearest-even
    return (unsigned short)(u >> 16);
}
__device__ __forceinline__ float bf2f(unsigned short h) {
    return __uint_as_float(((unsigned int)h) << 16);
}

__device__ __forceinline__ void store_val(unsigned short* o, long long i, float v) { o[i] = f2bf(v); }
__device__ __forceinline__ void store_val(float* o, long long i, float v)          { o[i] = v; }

// ---------------------------------------------------------------------------
// Layer 1: direct 3x3 conv (Cin=3) + bias + relu + fused 2x2 maxpool.
// in : x NCHW f32 (4,3,512,512); out: NHWC bf16 (4,256,256,64)
// ---------------------------------------------------------------------------
__global__ __launch_bounds__(256) void conv1_pool(
    const float* __restrict__ x, const float* __restrict__ Wt,
    const float* __restrict__ bias, unsigned short* __restrict__ out)
{
    long long tid = (long long)blockIdx.x * 256 + threadIdx.x; // 4*256*256*64
    int co = (int)(tid & 63);
    long long p = tid >> 6;
    int xo = (int)(p & 255); p >>= 8;
    int yo = (int)(p & 255);
    int b  = (int)(p >> 8);

    float w[27];
    #pragma unroll
    for (int i = 0; i < 27; ++i) w[i] = Wt[co * 27 + i];
    float bv = bias[co];

    float m = -INFINITY;
    #pragma unroll
    for (int dy = 0; dy < 2; ++dy)
      #pragma unroll
      for (int dx = 0; dx < 2; ++dx) {
        int yy = 2 * yo + dy, xx = 2 * xo + dx;
        float acc = bv;
        #pragma unroll
        for (int ci = 0; ci < 3; ++ci)
          #pragma unroll
          for (int r = 0; r < 3; ++r) {
            int y2 = yy + r - 1;
            if (y2 < 0 || y2 >= 512) continue;
            #pragma unroll
            for (int s = 0; s < 3; ++s) {
              int x2 = xx + s - 1;
              if (x2 < 0 || x2 >= 512) continue;
              acc += x[(((long long)b * 3 + ci) * 512 + y2) * 512 + x2] * w[ci * 9 + r * 3 + s];
            }
          }
        m = fmaxf(m, fmaxf(acc, 0.f));
      }
    out[tid] = f2bf(m);
}

// ---------------------------------------------------------------------------
// Pack OIHW f32 weights -> [cout][k] bf16 with k = (ky*3+kx)*Cin + ci.
// ---------------------------------------------------------------------------
template <int Cout, int Cin>
__global__ __launch_bounds__(256) void pack_weights(
    const float* __restrict__ Wsrc, unsigned short* __restrict__ dst)
{
    constexpr int K = 9 * Cin;
    long long tid = (long long)blockIdx.x * 256 + threadIdx.x;
    if (tid >= (long long)Cout * K) return;
    int co = (int)(tid / K);
    int k  = (int)(tid % K);
    int rs = k / Cin;
    int ci = k & (Cin - 1);
    int r = rs / 3, s = rs % 3;
    dst[tid] = f2bf(Wsrc[(((long long)co * Cin + ci) * 3 + r) * 3 + s]);
}

// ---------------------------------------------------------------------------
// Implicit-GEMM 3x3 SAME conv via v_wmma_f32_16x16x32_bf16.
//   M = 4*H*W pixels, N = Cout, K = 9*Cin (Cin % 32 == 0, power of two).
// Block = 128 threads = 4 waves. Wave tile = 16 (M) x 64 (N): the A fragment
// (scattered NHWC activation load) is loaded once per K-chunk and reused by
// 4 WMMAs; B fragments stream contiguously from packed [cout][k] weights.
// Block covers 64 M-rows x 64 N-cols. grid = (M/64, Cout/64).
//
// A fragment (16-bit A 16x32): lane<16 -> K {0..7,16..23}, lane>=16 -> +8.
// B fragment (16-bit B 32x16): lane<16 -> K {0..7,16..23}, lane>=16 -> +8,
//   column = lane&15.
// ---------------------------------------------------------------------------
template <int H, int W, int Cin, int Cout, typename OutT>
__global__ __launch_bounds__(128) void conv_wmma(
    const unsigned short* __restrict__ act,   // NHWC bf16
    const unsigned short* __restrict__ wpk,   // [Cout][K] bf16
    const float* __restrict__ bias,
    OutT* __restrict__ out)                   // NHWC
{
    constexpr int K  = 9 * Cin;
    constexpr int HW = H * W;

    const int lane = threadIdx.x & 31;
    const int wave = threadIdx.x >> 5;
    const int half = lane >> 4;
    const int l16  = lane & 15;

    const int m_base = (blockIdx.x * 4 + wave) * 16;
    const int m_row  = m_base + l16;          // A-matrix row for this lane
    const int bimg = m_row / HW;              // compile-time pow2 -> shifts
    const int rem  = m_row & (HW - 1);
    const int y = rem / W;
    const int x = rem & (W - 1);

    const int n_base = blockIdx.y * 64;
    const int ncol   = n_base + l16;          // column for B/C subtile 0

    v8f acc[4] = {};

    for (int k0 = 0; k0 < K; k0 += 32) {
        // Cin % 32 == 0 and k0 % 32 == 0 => whole chunk has one (ky,kx) tap.
        int rs    = k0 / Cin;
        int cbase = k0 & (Cin - 1);
        int r = rs / 3, s = rs % 3;
        int yy = y + r - 1;
        int xx = x + s - 1;

        Frag16 A;
        if ((unsigned)yy < (unsigned)H && (unsigned)xx < (unsigned)W) {
            const unsigned short* pa =
                act + ((((long long)bimg * H + yy) * W + xx) * Cin + cbase);
            A.q[0] = *(const U128*)(pa + half * 8);        // K: half*8   .. +7
            A.q[1] = *(const U128*)(pa + 16 + half * 8);   // K: 16+half*8.. +7
        } else {
            A.q[0] = U128{0u, 0u, 0u, 0u};
            A.q[1] = U128{0u, 0u, 0u, 0u};
        }

        #pragma unroll
        for (int nt = 0; nt < 4; ++nt) {
            Frag16 Bf;
            const unsigned short* pb =
                wpk + ((long long)(ncol + nt * 16) * K + k0 + half * 8);
            Bf.q[0] = *(const U128*)(pb);
            Bf.q[1] = *(const U128*)(pb + 16);
            acc[nt] = __builtin_amdgcn_wmma_f32_16x16x32_bf16(
                false, A.v, false, Bf.v, (short)0, acc[nt], false, false);
        }
    }

    // C/D layout: lane l16 = column N; VGPR rr = row rr (lanes<16) / rr+8.
    #pragma unroll
    for (int nt = 0; nt < 4; ++nt) {
        const int n = ncol + nt * 16;
        const float bv = bias[n];
        #pragma unroll
        for (int rr = 0; rr < 8; ++rr) {
            int m  = m_base + rr + half * 8;
            int bi = m / HW, rm = m & (HW - 1);
            int yo = rm / W, xo = rm & (W - 1);
            float v = fmaxf(acc[nt][rr] + bv, 0.f);
            long long oidx = (((long long)bi * H + yo) * W + xo) * Cout + n;
            store_val(out, oidx, v);
        }
    }
}

// ---------------------------------------------------------------------------
// 2x2 maxpool, NHWC bf16 -> NHWC bf16 (pow2 dims as templates: shift math).
// ---------------------------------------------------------------------------
template <int Ho, int Wo, int C>
__global__ __launch_bounds__(256) void pool2x2_bf16(
    const unsigned short* __restrict__ in, unsigned short* __restrict__ out)
{
    long long tid = (long long)blockIdx.x * 256 + threadIdx.x; // 4*Ho*Wo*C
    int c = (int)(tid & (C - 1));
    long long p = tid / C;
    int xo = (int)(p & (Wo - 1)); p /= Wo;
    int yo = (int)(p & (Ho - 1));
    int b  = (int)(p / Ho);
    constexpr int Hi = Ho * 2, Wi = Wo * 2;
    float m = -INFINITY;
    #pragma unroll
    for (int dy = 0; dy < 2; ++dy)
      #pragma unroll
      for (int dx = 0; dx < 2; ++dx)
        m = fmaxf(m, bf2f(in[(((long long)b * Hi + 2 * yo + dy) * Wi + (2 * xo + dx)) * C + c]));
    out[tid] = f2bf(m);
}

// ---------------------------------------------------------------------------
// Final 2x2 maxpool: NHWC f32 (4,64,64,512) -> NCHW f32 (4,512,32,32) = feat.
// ---------------------------------------------------------------------------
__global__ __launch_bounds__(256) void pool2x2_f32_nchw(
    const float* __restrict__ in, float* __restrict__ out)
{
    long long tid = (long long)blockIdx.x * 256 + threadIdx.x; // 2,097,152
    int xo = (int)(tid & 31); long long p = tid >> 5;
    int yo = (int)(p & 31); p >>= 5;
    int co = (int)(p & 511);
    int b  = (int)(p >> 9);
    float m = -INFINITY;
    #pragma unroll
    for (int dy = 0; dy < 2; ++dy)
      #pragma unroll
      for (int dx = 0; dx < 2; ++dx)
        m = fmaxf(m, in[(((long long)b * 64 + 2 * yo + dy) * 64 + 2 * xo + dx) * 512 + co]);
    out[tid] = m;
}

// ---------------------------------------------------------------------------
// Anchor targets. fh = fw = 32; 9 anchor types x 1024 cells = 9216 anchors.
// ---------------------------------------------------------------------------
__device__ __forceinline__ void anchor_box(int a, float& x1, float& y1,
                                           float& x2, float& y2)
{
    int t   = a >> 10;
    int pos = a & 1023;
    float gy = (float)(pos >> 5);
    float gx = (float)(pos & 31);
    int i = t / 3, j = t - 3 * i;
    float h = 2.f * (float)(j + 1);            // SCALES = {2,4,6}
    float w = 0.5f * (float)(i + 1) * h;       // ARS    = {0.5,1,1.5}
    x1 = fminf(fmaxf(gx - 0.5f * w, 0.5f), 32.f);
    y1 = fminf(fmaxf(gy - 0.5f * h, 0.5f), 32.f);
    x2 = fminf(fmaxf(gx + 0.5f * w, 0.5f), 32.f);
    y2 = fminf(fmaxf(gy + 0.5f * h, 0.5f), 32.f);
}

__device__ __forceinline__ void load_gt(const float* gp, float& x1, float& y1,
                                        float& x2, float& y2)
{
    x1 = (gp[0] == -1.f) ? -1.f : gp[0] * 32.f;
    y1 = (gp[1] == -1.f) ? -1.f : gp[1] * 32.f;
    x2 = (gp[2] == -1.f) ? -1.f : gp[2] * 32.f;
    y2 = (gp[3] == -1.f) ? -1.f : gp[3] * 32.f;
}

__device__ __forceinline__ float iou_calc(float ax1, float ay1, float ax2, float ay2,
                                          float aarea,
                                          float gx1, float gy1, float gx2, float gy2)
{
    float ix1 = fmaxf(ax1, gx1);
    float iy1 = fmaxf(ay1, gy1);
    float ix2 = fminf(ax2, gx2);
    float iy2 = fminf(ay2, gy2);
    float inter = fmaxf(ix2 - ix1, 0.f) * fmaxf(iy2 - iy1, 0.f);
    float garea = (gx2 - gx1) * (gy2 - gy1);
    float uni   = aarea + garea - inter;
    return inter / fmaxf(uni, 1e-8f);
}

// Pass 1: max IoU over all anchors for each (b, gt).
__global__ __launch_bounds__(256) void gt_max_kernel(
    const float* __restrict__ gts, float* __restrict__ maxpg)
{
    int b = blockIdx.x / 50, j = blockIdx.x % 50;
    const float* gp = gts + (long long)(b * 50 + j) * 5;
    float gx1, gy1, gx2, gy2;
    load_gt(gp, gx1, gy1, gx2, gy2);

    float best = -INFINITY;
    for (int a = threadIdx.x; a < 9216; a += 256) {
        float ax1, ay1, ax2, ay2;
        anchor_box(a, ax1, ay1, ax2, ay2);
        float aarea = (ax2 - ax1) * (ay2 - ay1);
        best = fmaxf(best, iou_calc(ax1, ay1, ax2, ay2, aarea, gx1, gy1, gx2, gy2));
    }
    __shared__ float red[256];
    red[threadIdx.x] = best;
    __syncthreads();
    for (int s = 128; s > 0; s >>= 1) {
        if (threadIdx.x < s) red[threadIdx.x] = fmaxf(red[threadIdx.x], red[threadIdx.x + s]);
        __syncthreads();
    }
    if (threadIdx.x == 0) maxpg[blockIdx.x] = red[0];
}

// Pass 2: per-anchor assignment + box regression targets.
__global__ __launch_bounds__(256) void targets_kernel(
    const float* __restrict__ gts, const float* __restrict__ maxpg,
    float* __restrict__ tout)
{
    int tid = blockIdx.x * 256 + threadIdx.x;
    if (tid >= 4 * 9216) return;
    int b = tid / 9216, a = tid - b * 9216;

    float ax1, ay1, ax2, ay2;
    anchor_box(a, ax1, ay1, ax2, ay2);
    float aarea = (ax2 - ax1) * (ay2 - ay1);

    float maxpa = -INFINITY;
    int arg = 0;
    bool pos = false;
    for (int j = 0; j < 50; ++j) {
        const float* gp = gts + (long long)(b * 50 + j) * 5;
        float gx1, gy1, gx2, gy2;
        load_gt(gp, gx1, gy1, gx2, gy2);
        float iou = iou_calc(ax1, ay1, ax2, ay2, aarea, gx1, gy1, gx2, gy2);
        if (iou > maxpa) { maxpa = iou; arg = j; }          // first-max argmax
        float mp = maxpg[b * 50 + j];
        pos = pos || ((iou == mp) && (mp > 0.f)) || (iou > 0.7f);
    }

    const float* gp = gts + (long long)(b * 50 + arg) * 5;
    float gx1, gy1, gx2, gy2;
    load_gt(gp, gx1, gy1, gx2, gy2);
    float cls = gp[4];

    float posf = pos ? 1.f : 0.f;
    float acx = (ax1 + ax2) * 0.5f, acy = (ay1 + ay2) * 0.5f;
    float aw = ax2 - ax1, ah = ay2 - ay1;
    float gcx = (gx1 + gx2) * 0.5f, gcy = (gy1 + gy2) * 0.5f;
    float gw = gx2 - gx1, gh = gy2 - gy1;
    float tx = (gcx - acx) / aw * posf;
    float ty = (gcy - acy) / ah * posf;
    float gws = pos ? gw : aw;
    float ghs = pos ? gh : ah;
    float tw = logf(gws / aw) * posf;
    float th = logf(ghs / ah) * posf;
    float neg = (maxpa < 0.3f) ? 1.f : 0.f;

    float* o = tout + (long long)tid * 8;
    o[0] = tx; o[1] = ty; o[2] = tw; o[3] = th;
    o[4] = maxpa * posf; o[5] = cls * posf; o[6] = posf; o[7] = neg;
}

// ---------------------------------------------------------------------------
// Launch
// ---------------------------------------------------------------------------
extern "C" void kernel_launch(void* const* d_in, const int* in_sizes, int n_in,
                              void* d_out, int out_size, void* d_ws, size_t ws_size,
                              hipStream_t stream)
{
    const float* x   = (const float*)d_in[0];
    const float* gts = (const float*)d_in[1];
    const float* W1  = (const float*)d_in[2]; const float* b1 = (const float*)d_in[3];
    const float* W2  = (const float*)d_in[4]; const float* b2 = (const float*)d_in[5];
    const float* W3  = (const float*)d_in[6]; const float* b3 = (const float*)d_in[7];
    const float* W4  = (const float*)d_in[8]; const float* b4 = (const float*)d_in[9];
    float* out = (float*)d_out;

    // Workspace carve (~104 MB total).
    char* ws = (char*)d_ws;
    unsigned short* bufA = (unsigned short*)ws;      // 67,108,864 B (largest pre-pool act)
    float*          bufAf = (float*)ws;              // alias: layer-4 f32 output (33.5 MB)
    char* p = ws + 67108864LL;
    unsigned short* bufB = (unsigned short*)p; p += 33554432LL;   // pooled acts (bf16 NHWC)
    unsigned short* wp2  = (unsigned short*)p; p += 2LL * 128 * 576;
    unsigned short* wp3  = (unsigned short*)p; p += 2LL * 256 * 1152;
    unsigned short* wp4  = (unsigned short*)p; p += 2LL * 512 * 2304;
    float* maxpg = (float*)p;                                     // 200 floats

    // Pack bf16 weights for WMMA layers.
    pack_weights<128, 64><<<(128 * 576 + 255) / 256, 256, 0, stream>>>(W2, wp2);
    pack_weights<256, 128><<<(256 * 1152 + 255) / 256, 256, 0, stream>>>(W3, wp3);
    pack_weights<512, 256><<<(512 * 2304 + 255) / 256, 256, 0, stream>>>(W4, wp4);

    // Layer 1: fused conv+relu+pool -> bufB (4,256,256,64) bf16 NHWC.
    conv1_pool<<<(4LL * 256 * 256 * 64) / 256, 256, 0, stream>>>(x, W1, b1, bufB);

    // Layer 2: WMMA conv (256x256, 64->128) -> bufA; pool -> bufB.
    {
        dim3 g((4 * 256 * 256) / 64, 128 / 64);
        conv_wmma<256, 256, 64, 128, unsigned short><<<g, 128, 0, stream>>>(bufB, wp2, b2, bufA);
    }
    pool2x2_bf16<128, 128, 128><<<(4LL * 128 * 128 * 128) / 256, 256, 0, stream>>>(bufA, bufB);

    // Layer 3: WMMA conv (128x128, 128->256) -> bufA; pool -> bufB.
    {
        dim3 g((4 * 128 * 128) / 64, 256 / 64);
        conv_wmma<128, 128, 128, 256, unsigned short><<<g, 128, 0, stream>>>(bufB, wp3, b3, bufA);
    }
    pool2x2_bf16<64, 64, 256><<<(4LL * 64 * 64 * 256) / 256, 256, 0, stream>>>(bufA, bufB);

    // Layer 4: WMMA conv (64x64, 256->512), f32 output -> bufAf; final pool -> feat.
    {
        dim3 g((4 * 64 * 64) / 64, 512 / 64);
        conv_wmma<64, 64, 256, 512, float><<<g, 128, 0, stream>>>(bufB, wp4, b4, bufAf);
    }
    pool2x2_f32_nchw<<<2097152 / 256, 256, 0, stream>>>(bufAf, out);

    // Anchor targets -> d_out after feat (offset 4*512*32*32 = 2,097,152 floats).
    gt_max_kernel<<<200, 256, 0, stream>>>(gts, maxpg);
    targets_kernel<<<(4 * 9216 + 255) / 256, 256, 0, stream>>>(gts, maxpg, out + 2097152);
}
